// SNNLanguageModel_84353157693918
// MI455X (gfx1250) — compile-verified, matrix operand order checked by reading, not run on hardware
//
#include <hip/hip_runtime.h>
#include <hip/hip_fp16.h>

typedef __attribute__((ext_vector_type(16))) _Float16 v16h;
typedef __attribute__((ext_vector_type(8)))  float    v8f;

#define V_    6144
#define D_    1024
#define KB    16
#define LL    4
#define DFF_  3072
#define BB    2
#define SS    64
#define DN_   4096
#define TT    (SS*KB)    /* 1024 timesteps */
#define MROWS (TT*BB)    /* 2048 GEMM rows */
#define VTHMIN 0.1f

// ---------------------------------------------------------------------------
// Epilogue modes for the generic WMMA GEMM
// ---------------------------------------------------------------------------
#define EPI_F32    0   // Out f32: C = (accum? C:0) + acc + bias
#define EPI_F16RAW 1   // Out f16: C = acc
#define EPI_F16SIG 2   // Out f16: C = sigmoid(acc + bias)
#define EPI_F16TH  3   // Out f16: C = max(softplus(acc + bias), 0.1)

__device__ __forceinline__ float sigf(float x) { return 1.0f / (1.0f + expf(-x)); }

#define WMMA_F16(a, b, c) \
  __builtin_amdgcn_wmma_f32_16x16x32_f16(false, (a), false, (b), (short)0, (c), false, false)

// Pack 16 f32 weights (two 8-float chunks at +0 and +16 elements)
// into a v16h B-fragment for v_wmma_f32_16x16x32_f16.
__device__ __forceinline__ v16h load_bfrag(const float* __restrict__ w0,
                                           const float* __restrict__ w1) {
  float t[16];
  *(float4*)&t[0]  = ((const float4*)w0)[0];
  *(float4*)&t[4]  = ((const float4*)w0)[1];
  *(float4*)&t[8]  = ((const float4*)w1)[0];
  *(float4*)&t[12] = ((const float4*)w1)[1];
  v16h b;
#pragma unroll
  for (int j = 0; j < 16; ++j) b[j] = (_Float16)t[j];
  return b;
}

// ---------------------------------------------------------------------------
// Generic GEMM: C(M,N) = A(M,K) x W(N,K)^T   (A f16 spikes, W f32 weights)
// Block = 256 threads = 8 waves; wave tile 32x32 (2x2 fragments, 4 WMMAs per
// K-step with full A/B fragment reuse); block tile 64x128.
// grid = (M/64, N/128). All dims divide exactly for this model.
// ---------------------------------------------------------------------------
__global__ void k_wmma_gemm(const _Float16* __restrict__ A,
                            const float*    __restrict__ W,
                            const float*    __restrict__ bias,
                            void*           __restrict__ Out,
                            int N, int Kd, int epi, int accum) {
  const int lane = threadIdx.x & 31;
  const int wave = threadIdx.x >> 5;
  const int m0 = blockIdx.x * 64  + (wave >> 2) * 32;
  const int n0 = blockIdx.y * 128 + (wave & 3) * 32;
  const int r    = lane & 15;          // A-row / B-col within 16-wide frag
  const int koff = (lane >> 4) * 8;    // half-wave K-chunk select

  const _Float16* a0 = A + (size_t)(m0 + r) * Kd + koff;
  const _Float16* a1 = a0 + (size_t)16 * Kd;
  const float*    b0 = W + (size_t)(n0 + r) * Kd + koff;
  const float*    b1 = b0 + (size_t)16 * Kd;

  v8f acc00 = {}, acc01 = {}, acc10 = {}, acc11 = {};
  for (int k0 = 0; k0 < Kd; k0 += 32) {
    union { v16h v; uint4 q[2]; } A0, A1;
    A0.q[0] = *(const uint4*)(a0 + k0);
    A0.q[1] = *(const uint4*)(a0 + k0 + 16);
    A1.q[0] = *(const uint4*)(a1 + k0);
    A1.q[1] = *(const uint4*)(a1 + k0 + 16);
    v16h B0 = load_bfrag(b0 + k0, b0 + k0 + 16);
    v16h B1 = load_bfrag(b1 + k0, b1 + k0 + 16);
    __builtin_prefetch(b0 + k0 + 64, 0, 1);   // global_prefetch_b8 on weights
    __builtin_prefetch(b1 + k0 + 64, 0, 1);
    acc00 = WMMA_F16(A0.v, B0, acc00);
    acc01 = WMMA_F16(A0.v, B1, acc01);
    acc10 = WMMA_F16(A1.v, B0, acc10);
    acc11 = WMMA_F16(A1.v, B1, acc11);
  }

  const int nn    = lane & 15;
  const int mbase = (lane >> 4) * 8;
  v8f accs[2][2] = {{acc00, acc01}, {acc10, acc11}};

#pragma unroll
  for (int mi = 0; mi < 2; ++mi) {
#pragma unroll
    for (int ni = 0; ni < 2; ++ni) {
      const int col = n0 + ni * 16 + nn;
      const size_t base = (size_t)(m0 + mi * 16 + mbase) * N + col;
      const float bv = bias ? bias[col] : 0.0f;
      if (epi == EPI_F32) {
        float* C = (float*)Out;
#pragma unroll
        for (int j = 0; j < 8; ++j) {
          size_t idx = base + (size_t)j * N;
          float prev = accum ? C[idx] : 0.0f;
          C[idx] = prev + accs[mi][ni][j] + bv;
        }
      } else {
        _Float16* C = (_Float16*)Out;
#pragma unroll
        for (int j = 0; j < 8; ++j) {
          size_t idx = base + (size_t)j * N;
          float x = accs[mi][ni][j] + bv;
          if (epi == EPI_F16SIG) {
            x = sigf(x);
          } else if (epi == EPI_F16TH) {
            float sp = (x > 20.0f) ? x : log1pf(expf(x));
            x = fmaxf(sp, VTHMIN);
          }
          C[idx] = (_Float16)x;
        }
      }
    }
  }
}

// ---------------------------------------------------------------------------
// Dual GEMM sharing the A-operand: u = A@W0^T ; alpha = sigmoid(A@W1^T + b1)
// Out f16 = alpha * u.  Wave tile 32(M)x16(N) x two weight matrices
// (4 WMMAs per K-step, A fragments reused).  Block tile 64x64.
// ---------------------------------------------------------------------------
__global__ void k_wmma_gemm_dual(const _Float16* __restrict__ A,
                                 const float*    __restrict__ W0,
                                 const float*    __restrict__ W1,
                                 const float*    __restrict__ b1,
                                 _Float16*       __restrict__ Out,
                                 int N, int Kd) {
  const int lane = threadIdx.x & 31;
  const int wave = threadIdx.x >> 5;
  const int m0 = blockIdx.x * 64 + (wave >> 2) * 32;
  const int n0 = blockIdx.y * 64 + (wave & 3) * 16;
  const int r    = lane & 15;
  const int koff = (lane >> 4) * 8;

  const _Float16* a0  = A  + (size_t)(m0 + r) * Kd + koff;
  const _Float16* a1  = a0 + (size_t)16 * Kd;
  const float*    w0r = W0 + (size_t)(n0 + r) * Kd + koff;
  const float*    w1r = W1 + (size_t)(n0 + r) * Kd + koff;

  v8f accU0 = {}, accU1 = {}, accA0 = {}, accA1 = {};
  for (int k0 = 0; k0 < Kd; k0 += 32) {
    union { v16h v; uint4 q[2]; } A0, A1;
    A0.q[0] = *(const uint4*)(a0 + k0);
    A0.q[1] = *(const uint4*)(a0 + k0 + 16);
    A1.q[0] = *(const uint4*)(a1 + k0);
    A1.q[1] = *(const uint4*)(a1 + k0 + 16);
    v16h B0 = load_bfrag(w0r + k0, w0r + k0 + 16);
    v16h B1 = load_bfrag(w1r + k0, w1r + k0 + 16);
    __builtin_prefetch(w0r + k0 + 64, 0, 1);
    __builtin_prefetch(w1r + k0 + 64, 0, 1);
    accU0 = WMMA_F16(A0.v, B0, accU0);
    accU1 = WMMA_F16(A1.v, B0, accU1);
    accA0 = WMMA_F16(A0.v, B1, accA0);
    accA1 = WMMA_F16(A1.v, B1, accA1);
  }

  const int nn    = lane & 15;
  const int mbase = (lane >> 4) * 8;
  const float bv = b1[n0 + nn];
  v8f accU[2] = {accU0, accU1};
  v8f accA[2] = {accA0, accA1};
#pragma unroll
  for (int mi = 0; mi < 2; ++mi) {
    const size_t base = (size_t)(m0 + mi * 16 + mbase) * N + n0 + nn;
#pragma unroll
    for (int j = 0; j < 8; ++j) {
      float al = sigf(accA[mi][j] + bv);
      Out[base + (size_t)j * N] = (_Float16)(al * accU[mi][j]);
    }
  }
}

// ---------------------------------------------------------------------------
// Encode: token embedding -> fp16 bit-plane spikes h(T,B,D), MSB first
// ---------------------------------------------------------------------------
__global__ void k_encode(const int* __restrict__ tok,
                         const float* __restrict__ embed,
                         float* __restrict__ h) {
  int i = blockIdx.x * blockDim.x + threadIdx.x;   // over B*S*D
  int d = i % D_;
  int s = (i / D_) % SS;
  int b = i / (D_ * SS);
  int t0 = tok[b * SS + s];
  float e = embed[(size_t)t0 * D_ + d];
  unsigned short bits = __half_as_ushort(__float2half(e));
#pragma unroll
  for (int k = 0; k < KB; ++k) {
    float sp = (float)((bits >> (15 - k)) & 1);
    int t = s * KB + k;
    h[((size_t)t * BB + b) * D_ + d] = sp;
  }
}

// ---------------------------------------------------------------------------
// PLIF scan, f32 input -> f16 spikes.  One thread per (b,channel) chain.
// v[t] = sigmoid(w[c]) * v[t-1] + x[t];  spike = v >= vth[c]
// ---------------------------------------------------------------------------
__global__ void k_plif_f32(const float* __restrict__ x, _Float16* __restrict__ out,
                           const float* __restrict__ w, const float* __restrict__ vth,
                           int C, int BC) {
  int tid = blockIdx.x * blockDim.x + threadIdx.x;
  if (tid >= BC) return;
  int c = tid % C;
  float tau = sigf(w[c]);
  float th  = vth[c];
  float v = 0.0f;
  size_t idx = tid;
  for (int t = 0; t < TT; ++t, idx += BC) {
    v = tau * v + x[idx];
    out[idx] = (v >= th) ? (_Float16)1.0f : (_Float16)0.0f;
  }
}

// Same scan, f16 input, in-place spike output.
__global__ void k_plif_f16(_Float16* __restrict__ x,
                           const float* __restrict__ w, const float* __restrict__ vth,
                           int C, int BC) {
  int tid = blockIdx.x * blockDim.x + threadIdx.x;
  if (tid >= BC) return;
  int c = tid % C;
  float tau = sigf(w[c]);
  float th  = vth[c];
  float v = 0.0f;
  size_t idx = tid;
  for (int t = 0; t < TT; ++t, idx += BC) {
    v = tau * v + (float)x[idx];
    x[idx] = (v >= th) ? (_Float16)1.0f : (_Float16)0.0f;
  }
}

// ---------------------------------------------------------------------------
// SNN-block neuron scan with input-dependent decay/threshold/gate.
// au buffer is read (alpha*u) and overwritten with spike*gate.
// ---------------------------------------------------------------------------
__global__ void k_scan_block(_Float16* __restrict__ au,
                             const _Float16* __restrict__ beta,
                             const _Float16* __restrict__ th,
                             const _Float16* __restrict__ g,
                             int BC) {
  int tid = blockIdx.x * blockDim.x + threadIdx.x;
  if (tid >= BC) return;
  float v = 0.0f;
  size_t idx = tid;
  for (int t = 0; t < TT; ++t, idx += BC) {
    float bt = (float)beta[idx];
    float a  = (float)au[idx];
    v = bt * v + a;
    bool s = (v >= (float)th[idx]);
    au[idx] = s ? g[idx] : (_Float16)0.0f;
  }
}

// a *= b elementwise (g2 * u2 for the spiking SwiGLU)
__global__ void k_mul16(_Float16* __restrict__ a, const _Float16* __restrict__ b,
                        unsigned int n) {
  unsigned int i = blockIdx.x * blockDim.x + threadIdx.x;
  if (i < n) a[i] = a[i] * b[i];
}

// Decode: mean over K bit-plane timesteps -> dec16(B*S, D) f16
__global__ void k_decode(const float* __restrict__ h, _Float16* __restrict__ dec16) {
  int i = blockIdx.x * blockDim.x + threadIdx.x;  // over B*S*D
  int d = i % D_;
  int s = (i / D_) % SS;
  int b = i / (D_ * SS);
  float acc = 0.0f;
#pragma unroll
  for (int k = 0; k < KB; ++k)
    acc += h[((size_t)(s * KB + k) * BB + b) * D_ + d];
  dec16[((size_t)b * SS + s) * D_ + d] = (_Float16)(acc * (1.0f / KB));
}

// Gain-only layernorm (LateralInhibition), one block per (b,s) row, f16 out.
__global__ void k_ln_gain(const float* __restrict__ y, const float* __restrict__ gain,
                          _Float16* __restrict__ out) {
  __shared__ float red[256];
  const int row = blockIdx.x;
  const int tid = threadIdx.x;
  const float* yr = y + (size_t)row * D_;
  float s = 0.0f;
  for (int d = tid; d < D_; d += 256) s += yr[d];
  red[tid] = s; __syncthreads();
  for (int o = 128; o > 0; o >>= 1) { if (tid < o) red[tid] += red[tid + o]; __syncthreads(); }
  float mu = red[0] / D_;
  __syncthreads();
  float s2 = 0.0f;
  for (int d = tid; d < D_; d += 256) { float t = yr[d] - mu; s2 += t * t; }
  red[tid] = s2; __syncthreads();
  for (int o = 128; o > 0; o >>= 1) { if (tid < o) red[tid] += red[tid + o]; __syncthreads(); }
  float inv = rsqrtf(red[0] / D_ + 1e-6f);
  for (int d = tid; d < D_; d += 256)
    out[(size_t)row * D_ + d] = (_Float16)(gain[d] * (yr[d] - mu) * inv);
}

// ---------------------------------------------------------------------------
// Host-side orchestration
// ---------------------------------------------------------------------------
extern "C" void kernel_launch(void* const* d_in, const int* in_sizes, int n_in,
                              void* d_out, int out_size, void* d_ws, size_t ws_size,
                              hipStream_t stream) {
  (void)in_sizes; (void)n_in; (void)out_size; (void)ws_size;
  const int*   tok       = (const int*)  d_in[0];
  const float* embed     = (const float*)d_in[1];
  const float* gain      = (const float*)d_in[2];
  const float* decode_w  = (const float*)d_in[3];
  const float* decode_b  = (const float*)d_in[4];
  const float* in1_w     = (const float*)d_in[5];
  const float* in1_vth   = (const float*)d_in[6];
  const float* in2_w     = (const float*)d_in[7];
  const float* in2_vth   = (const float*)d_in[8];
  const float* W_in      = (const float*)d_in[9];
  const float* W_beta    = (const float*)d_in[10];
  const float* b_beta    = (const float*)d_in[11];
  const float* W_alpha   = (const float*)d_in[12];
  const float* b_alpha   = (const float*)d_in[13];
  const float* W_th      = (const float*)d_in[14];
  const float* b_th      = (const float*)d_in[15];
  const float* W_gate    = (const float*)d_in[16];
  const float* W_skip    = (const float*)d_in[17];
  const float* W_out     = (const float*)d_in[18];
  const float* outn_w    = (const float*)d_in[19];
  const float* outn_vth  = (const float*)d_in[20];
  const float* bproj     = (const float*)d_in[21];
  const float* ffn_gate  = (const float*)d_in[22];
  const float* ffn_up    = (const float*)d_in[23];
  const float* ffn_down  = (const float*)d_in[24];
  const float* ffn_skip  = (const float*)d_in[25];
  const float* gn_w      = (const float*)d_in[26];
  const float* gn_vth    = (const float*)d_in[27];
  const float* un_w      = (const float*)d_in[28];
  const float* un_vth    = (const float*)d_in[29];
  const float* fon_w     = (const float*)d_in[30];
  const float* fon_vth   = (const float*)d_in[31];
  const float* ffn_proj  = (const float*)d_in[32];

  // Workspace carve-out (~118 MB)
  char* ws = (char*)d_ws;
  size_t off = 0;
  auto carve = [&](size_t bytes) -> void* {
    void* p = ws + off;
    off += (bytes + 255) & ~(size_t)255;
    return p;
  };
  float*     h       = (float*)    carve((size_t)TT * BB * D_   * 4); // residual stream
  _Float16*  spk     = (_Float16*) carve((size_t)TT * BB * D_   * 2); // s1 / s2 spikes
  _Float16*  blkspk  = (_Float16*) carve((size_t)TT * BB * D_   * 2); // blk / f spikes
  float*     blkf    = (float*)    carve((size_t)TT * BB * D_   * 4); // blk / f pre-act
  _Float16*  beta16  = (_Float16*) carve((size_t)TT * BB * DN_  * 2);
  _Float16*  au16    = (_Float16*) carve((size_t)TT * BB * DN_  * 2); // alpha*u -> s*g
  _Float16*  th16    = (_Float16*) carve((size_t)TT * BB * DN_  * 2);
  _Float16*  g16     = (_Float16*) carve((size_t)TT * BB * DN_  * 2);
  _Float16*  gpre    = (_Float16*) carve((size_t)TT * BB * DFF_ * 2); // gate pre -> g2 -> g2*u2
  _Float16*  upre    = (_Float16*) carve((size_t)TT * BB * DFF_ * 2); // up pre   -> u2
  _Float16*  dec16   = (_Float16*) carve((size_t)BB * SS * D_   * 2);
  float*     yf      = (float*)    carve((size_t)BB * SS * D_   * 4);
  _Float16*  y16     = (_Float16*) carve((size_t)BB * SS * D_   * 2);

  const dim3 blk256(256);
  const dim3 gN4096(MROWS / 64, DN_  / 128);   // (32, 32)
  const dim3 gN1024(MROWS / 64, D_   / 128);   // (32, 8)
  const dim3 gN3072(MROWS / 64, DFF_ / 128);   // (32, 24)
  const dim3 gDual (MROWS / 64, DN_  / 64);    // (32, 64)
  const dim3 gDec  ((BB * SS) / 64, D_ / 128); // (2, 8)
  const dim3 gHead ((BB * SS) / 64, V_ / 128); // (2, 48)

  // 1. bit-plane encode
  k_encode<<<(BB * SS * D_) / 256, blk256, 0, stream>>>(tok, embed, h);

  for (int i = 0; i < LL; ++i) {
    const float* Wi_in   = W_in    + (size_t)i * DN_ * D_;
    const float* Wi_be   = W_beta  + (size_t)i * DN_ * D_;
    const float* Wi_al   = W_alpha + (size_t)i * DN_ * D_;
    const float* Wi_th   = W_th    + (size_t)i * DN_ * D_;
    const float* Wi_ga   = W_gate  + (size_t)i * DN_ * D_;
    const float* Wi_sk   = W_skip  + (size_t)i * D_  * D_;
    const float* Wi_out  = W_out   + (size_t)i * D_  * DN_;
    const float* Wi_bp   = bproj   + (size_t)i * D_  * D_;
    const float* Wi_fg   = ffn_gate+ (size_t)i * DFF_* D_;
    const float* Wi_fu   = ffn_up  + (size_t)i * DFF_* D_;
    const float* Wi_fd   = ffn_down+ (size_t)i * D_  * DFF_;
    const float* Wi_fs   = ffn_skip+ (size_t)i * D_  * D_;
    const float* Wi_fp   = ffn_proj+ (size_t)i * D_  * D_;

    // --- SNN block ---
    k_plif_f32<<<(BB * D_) / 256, blk256, 0, stream>>>(
        h, spk, in1_w + i * D_, in1_vth + i * D_, D_, BB * D_);
    // u & alpha fused (shared A loads), au16 = sigmoid(xa+ba)*u
    k_wmma_gemm_dual<<<gDual, blk256, 0, stream>>>(
        spk, Wi_in, Wi_al, b_alpha + (size_t)i * DN_, au16, DN_, D_);
    k_wmma_gemm<<<gN4096, blk256, 0, stream>>>(
        spk, Wi_be, b_beta + (size_t)i * DN_, beta16, DN_, D_, EPI_F16SIG, 0);
    k_wmma_gemm<<<gN4096, blk256, 0, stream>>>(
        spk, Wi_th, b_th + (size_t)i * DN_, th16, DN_, D_, EPI_F16TH, 0);
    k_wmma_gemm<<<gN4096, blk256, 0, stream>>>(
        spk, Wi_ga, nullptr, g16, DN_, D_, EPI_F16SIG, 0);
    // neuron state scan; au16 <- spike * gate
    k_scan_block<<<(BB * DN_) / 256, blk256, 0, stream>>>(au16, beta16, th16, g16, BB * DN_);
    // blk = (s*g)@W_out^T + s1@W_skip^T
    k_wmma_gemm<<<gN1024, blk256, 0, stream>>>(
        au16, Wi_out, nullptr, blkf, D_, DN_, EPI_F32, 0);
    k_wmma_gemm<<<gN1024, blk256, 0, stream>>>(
        spk, Wi_sk, nullptr, blkf, D_, D_, EPI_F32, 1);
    k_plif_f32<<<(BB * D_) / 256, blk256, 0, stream>>>(
        blkf, blkspk, outn_w + i * D_, outn_vth + i * D_, D_, BB * D_);
    // h += blk_spike @ block_out_proj^T
    k_wmma_gemm<<<gN1024, blk256, 0, stream>>>(
        blkspk, Wi_bp, nullptr, h, D_, D_, EPI_F32, 1);

    // --- spiking SwiGLU FFN ---
    k_plif_f32<<<(BB * D_) / 256, blk256, 0, stream>>>(
        h, spk, in2_w + i * D_, in2_vth + i * D_, D_, BB * D_);
    k_wmma_gemm<<<gN3072, blk256, 0, stream>>>(
        spk, Wi_fg, nullptr, gpre, DFF_, D_, EPI_F16RAW, 0);
    k_wmma_gemm<<<gN3072, blk256, 0, stream>>>(
        spk, Wi_fu, nullptr, upre, DFF_, D_, EPI_F16RAW, 0);
    k_plif_f16<<<(BB * DFF_) / 256, blk256, 0, stream>>>(
        gpre, gn_w + (size_t)i * DFF_, gn_vth + (size_t)i * DFF_, DFF_, BB * DFF_);
    k_plif_f16<<<(BB * DFF_) / 256, blk256, 0, stream>>>(
        upre, un_w + (size_t)i * DFF_, un_vth + (size_t)i * DFF_, DFF_, BB * DFF_);
    k_mul16<<<(TT * BB * DFF_) / 256, blk256, 0, stream>>>(
        gpre, upre, (unsigned int)(TT * BB * DFF_));
    // f = (g2*u2)@ffn_down^T + s2@ffn_skip^T
    k_wmma_gemm<<<gN1024, blk256, 0, stream>>>(
        gpre, Wi_fd, nullptr, blkf, D_, DFF_, EPI_F32, 0);
    k_wmma_gemm<<<gN1024, blk256, 0, stream>>>(
        spk, Wi_fs, nullptr, blkf, D_, D_, EPI_F32, 1);
    k_plif_f32<<<(BB * D_) / 256, blk256, 0, stream>>>(
        blkf, blkspk, fon_w + i * D_, fon_vth + i * D_, D_, BB * D_);
    k_wmma_gemm<<<gN1024, blk256, 0, stream>>>(
        blkspk, Wi_fp, nullptr, h, D_, D_, EPI_F32, 1);
  }

  // decode + projection + gain-layernorm + tied LM head
  k_decode<<<(BB * SS * D_) / 256, blk256, 0, stream>>>(h, dec16);
  k_wmma_gemm<<<gDec, blk256, 0, stream>>>(
      dec16, decode_w, decode_b, yf, D_, D_, EPI_F32, 0);
  k_ln_gain<<<BB * SS, blk256, 0, stream>>>(yf, gain, y16);
  k_wmma_gemm<<<gHead, blk256, 0, stream>>>(
      y16, embed, nullptr, (float*)d_out, V_, D_, EPI_F32, 0);
}
